// STMambaEncoder_17781164605874
// MI455X (gfx1250) — compile-verified, matrix operand order checked by reading, not run on hardware
//
#include <hip/hip_runtime.h>
#include <hip/hip_bf16.h>

#define BB      4
#define LSEQ    257
#define RTOT    (BB*LSEQ)      // 1028 rows
#define RPAD    1040           // RTOT padded to 16
#define DMODEL  384
#define DINNER  768
#define NST     16
#define DTRANK  24
#define XDB     56             // DTR + 2N
#define KCONV   4
#define NTOKC   256
#define POSC    128
#define PATCHK  256            // C*P*P

typedef __attribute__((ext_vector_type(16))) _Float16 v16h;
typedef __attribute__((ext_vector_type(8)))  _Float16 v8h;
typedef __attribute__((ext_vector_type(8)))  float    v8f;

// ---------------------------------------------------------------------------
// WMMA GEMM on pre-padded f16 operands:
//   C[M,N] = A16[Mp,Kp] * W16[Np,Kp]^T (+bias[n]) (+softplus)
// A16 ld = Kp (mult of 32), W16 ld = Kp, Np mult of 64.
// One wave computes a 16(M) x 64(N) strip: A fragment loaded once per K-step,
// 4 v_wmma_f32_16x16x32_f16 issued on it. No conditionals in the K loop.
// ---------------------------------------------------------------------------
__global__ void gemm_f16wmma(const _Float16* __restrict__ A16,
                             const _Float16* __restrict__ W16,
                             const float* __restrict__ bias,
                             float* __restrict__ C, int ldc,
                             int M, int N, int Kp, int Np, int act)
{
    const int ntNg = Np >> 6;                       // groups of 4 N-tiles
    const int ntM  = (M + 15) >> 4;
    const int wid  = blockIdx.x * blockDim.y + threadIdx.y;
    if (wid >= ntM * ntNg) return;                  // wave-uniform exit
    const int tM = wid / ntNg;
    const int g  = wid % ntNg;

    const int lane = threadIdx.x;                   // 0..31 (wave32)
    const int lr   = lane & 15;
    const int kh   = (lane >> 4) << 3;              // 0 or 8 (K-half select)

    const _Float16* __restrict__ arow = A16 + (size_t)(tM * 16 + lr) * Kp + kh;
    const _Float16* __restrict__ br0  = W16 + (size_t)(g * 64 + lr) * Kp + kh;
    const size_t bstride = (size_t)16 * Kp;         // next N-tile of W

    v8f acc0 = {}, acc1 = {}, acc2 = {}, acc3 = {};
    for (int k0 = 0; k0 < Kp; k0 += 32) {
        v8h a0 = *(const v8h*)(arow + k0);
        v8h a1 = *(const v8h*)(arow + k0 + 16);
        v16h af = __builtin_shufflevector(a0, a1, 0,1,2,3,4,5,6,7,8,9,10,11,12,13,14,15);

        v8h b0, b1; v16h bf;
        b0 = *(const v8h*)(br0 + k0);
        b1 = *(const v8h*)(br0 + k0 + 16);
        bf = __builtin_shufflevector(b0, b1, 0,1,2,3,4,5,6,7,8,9,10,11,12,13,14,15);
        acc0 = __builtin_amdgcn_wmma_f32_16x16x32_f16(false, af, false, bf, (short)0, acc0, false, false);

        b0 = *(const v8h*)(br0 + bstride + k0);
        b1 = *(const v8h*)(br0 + bstride + k0 + 16);
        bf = __builtin_shufflevector(b0, b1, 0,1,2,3,4,5,6,7,8,9,10,11,12,13,14,15);
        acc1 = __builtin_amdgcn_wmma_f32_16x16x32_f16(false, af, false, bf, (short)0, acc1, false, false);

        b0 = *(const v8h*)(br0 + 2 * bstride + k0);
        b1 = *(const v8h*)(br0 + 2 * bstride + k0 + 16);
        bf = __builtin_shufflevector(b0, b1, 0,1,2,3,4,5,6,7,8,9,10,11,12,13,14,15);
        acc2 = __builtin_amdgcn_wmma_f32_16x16x32_f16(false, af, false, bf, (short)0, acc2, false, false);

        b0 = *(const v8h*)(br0 + 3 * bstride + k0);
        b1 = *(const v8h*)(br0 + 3 * bstride + k0 + 16);
        bf = __builtin_shufflevector(b0, b1, 0,1,2,3,4,5,6,7,8,9,10,11,12,13,14,15);
        acc3 = __builtin_amdgcn_wmma_f32_16x16x32_f16(false, af, false, bf, (short)0, acc3, false, false);
    }

    // C/D layout: n = lane%16, m = j + 8*(lane/16)
    const int mbase = tM * 16 + ((lane >> 4) << 3);
    v8f accs[4] = {acc0, acc1, acc2, acc3};
#pragma unroll
    for (int q = 0; q < 4; ++q) {
        const int cn = g * 64 + q * 16 + (lane & 15);
        if (cn >= N) continue;
        const float bv = bias ? bias[cn] : 0.f;
#pragma unroll
        for (int j = 0; j < 8; ++j) {
            const int cm = mbase + j;
            if (cm < M) {
                float v = accs[q][j] + bv;
                if (act == 1) v = (v > 20.f) ? v : log1pf(expf(v));  // softplus
                C[(size_t)cm * ldc + cn] = v;
            }
        }
    }
}

// ---------------------------------------------------------------------------
// f32 -> f16 convert with zero padding: dst[rowsP, colsP], ld = colsP
// ---------------------------------------------------------------------------
__global__ void cvt_pad_f16(const float* __restrict__ src, int rows, int cols,
                            int ldsrc, _Float16* __restrict__ dst,
                            int rowsP, int colsP)
{
    int idx = blockIdx.x * blockDim.x + threadIdx.x;
    if (idx >= rowsP * colsP) return;
    int r = idx / colsP, c = idx % colsP;
    dst[idx] = (r < rows && c < cols) ? (_Float16)src[(size_t)r * ldsrc + c]
                                      : (_Float16)0.f;
}

__global__ void zero_h16(_Float16* __restrict__ p, int n)
{
    int idx = blockIdx.x * blockDim.x + threadIdx.x;
    if (idx < n) p[idx] = (_Float16)0.f;
}

// ---------------------------------------------------------------------------
// Patchify: depth_seq (B,T,C,H,W) -> xr16 (B*NTOK, 256) f16
// ---------------------------------------------------------------------------
__global__ void gather_xr(const float* __restrict__ ds, _Float16* __restrict__ xr)
{
    int idx = blockIdx.x * blockDim.x + threadIdx.x;
    if (idx >= BB * NTOKC * PATCHK) return;
    int b   = idx / (NTOKC * PATCHK);
    int rem = idx % (NTOKC * PATCHK);
    int t   = rem / PATCHK;
    int p   = rem % PATCHK;
    int ph = p >> 4, pw = p & 15;
    int gh = t >> 3, gw = t & 7;           // GW = 8
    int r = gh * 16 + ph, c = gw * 16 + pw;
    xr[idx] = (_Float16)ds[(size_t)b * (512 * 128) + r * 128 + c];
}

// ---------------------------------------------------------------------------
// Insert cls at POS, add positional embedding -> hidden (B*L, 384) f32
// ---------------------------------------------------------------------------
__global__ void assemble_tokens(const float* __restrict__ tokG,
                                const float* __restrict__ cls,
                                const float* __restrict__ pos,
                                float* __restrict__ hidden)
{
    int idx = blockIdx.x * blockDim.x + threadIdx.x;
    if (idx >= RTOT * DMODEL) return;
    int b   = idx / (LSEQ * DMODEL);
    int rem = idx % (LSEQ * DMODEL);
    int l   = rem / DMODEL;
    int d   = rem % DMODEL;
    float tv;
    if (l < POSC)        tv = tokG[((size_t)b * NTOKC + l) * DMODEL + d];
    else if (l == POSC)  tv = cls[d];
    else                 tv = tokG[((size_t)b * NTOKC + l - 1) * DMODEL + d];
    hidden[idx] = tv + pos[l * DMODEL + d];
}

// ---------------------------------------------------------------------------
// residual = hidden (+ residual); LN -> hout (f32) and hout16 (f16, ld 384)
// One wave32 per row (D=384 -> 12 elems/lane), shuffle reductions.
// ---------------------------------------------------------------------------
__global__ void resid_ln(const float* __restrict__ hin, float* __restrict__ res,
                         float* __restrict__ hout, _Float16* __restrict__ hout16,
                         const float* __restrict__ w, const float* __restrict__ b,
                         int addres, int rows)
{
    int row = blockIdx.x * blockDim.y + threadIdx.y;
    if (row >= rows) return;
    int lane = threadIdx.x;
    const float* hr = hin + (size_t)row * DMODEL;
    float* rr = res + (size_t)row * DMODEL;
    float vals[12];
    float s = 0.f;
#pragma unroll
    for (int j = 0; j < 12; ++j) {
        int d = lane + j * 32;
        float v = hr[d] + (addres ? rr[d] : 0.f);
        vals[j] = v; s += v;
    }
#pragma unroll
    for (int o = 16; o >= 1; o >>= 1) s += __shfl_xor(s, o, 32);
    float mean = s * (1.f / DMODEL);
    float vs = 0.f;
#pragma unroll
    for (int j = 0; j < 12; ++j) { float c = vals[j] - mean; vs += c * c; }
#pragma unroll
    for (int o = 16; o >= 1; o >>= 1) vs += __shfl_xor(vs, o, 32);
    float rstd = rsqrtf(vs * (1.f / DMODEL) + 1e-5f);
#pragma unroll
    for (int j = 0; j < 12; ++j) {
        int d = lane + j * 32;
        rr[d] = vals[j];
        float o = (vals[j] - mean) * rstd * w[d] + b[d];
        hout[(size_t)row * DMODEL + d] = o;
        hout16[(size_t)row * DMODEL + d] = (_Float16)o;
    }
}

// ---------------------------------------------------------------------------
// Depthwise causal conv (K=4) + bias + SiLU; rev flips sequence per batch.
// Writes f32 (for the scan) and padded f16 (for the xproj GEMM).
// ---------------------------------------------------------------------------
__global__ void conv_silu(const float* __restrict__ x, int ldx,
                          const float* __restrict__ w, const float* __restrict__ bias,
                          float* __restrict__ xc, _Float16* __restrict__ xc16, int rev)
{
    int idx = blockIdx.x * blockDim.x + threadIdx.x;
    if (idx >= RTOT * DINNER) return;
    int b   = idx / (LSEQ * DINNER);
    int rem = idx % (LSEQ * DINNER);
    int l   = rem / DINNER;
    int d   = rem % DINNER;
    float acc = bias[d];
#pragma unroll
    for (int k = 0; k < KCONV; ++k) {
        int src = l + k - (KCONV - 1);
        if (src >= 0) {
            int sr = rev ? (LSEQ - 1 - src) : src;
            acc += x[((size_t)b * LSEQ + sr) * ldx + d] * w[d * KCONV + k];
        }
    }
    float s = acc / (1.f + expf(-acc));            // silu
    size_t o = ((size_t)b * LSEQ + l) * DINNER + d;
    xc[o]   = s;
    xc16[o] = (_Float16)s;
}

// ---------------------------------------------------------------------------
// Selective-scan: one thread per (b, d), N=16 state in registers.
// B/C vectors for each timestep staged in LDS. z gated (flipped if rev).
// ---------------------------------------------------------------------------
__global__ void ssm_scan(const float* __restrict__ xc, const float* __restrict__ dt,
                         const float* __restrict__ xdb, const float* __restrict__ z,
                         int ldz,
                         const float* __restrict__ Alog, const float* __restrict__ Dv,
                         float* __restrict__ y, int rev)
{
    const int b = blockIdx.y;
    const int d = blockIdx.x * blockDim.x + threadIdx.x;   // < DINNER
    __shared__ float sB[NST], sC[NST];
    float a[NST], h[NST];
#pragma unroll
    for (int n = 0; n < NST; ++n) { a[n] = -expf(Alog[d * NST + n]); h[n] = 0.f; }
    const float Dd = Dv[d];

    for (int t = 0; t < LSEQ; ++t) {
        const size_t row = (size_t)b * LSEQ + t;
        if (threadIdx.x < NST)            sB[threadIdx.x]       = xdb[row * XDB + DTRANK + threadIdx.x];
        else if (threadIdx.x < 2 * NST)   sC[threadIdx.x - NST] = xdb[row * XDB + DTRANK + threadIdx.x];
        __syncthreads();

        const float dtv = dt[row * DINNER + d];
        const float xv  = xc[row * DINNER + d];
        const size_t zrow = (size_t)b * LSEQ + (rev ? (LSEQ - 1 - t) : t);
        const float zv  = z[zrow * ldz + d];

        if (t + 1 < LSEQ) {                        // pull next step into cache
            __builtin_prefetch(dt + (row + 1) * DINNER + d, 0, 1);
            __builtin_prefetch(xc + (row + 1) * DINNER + d, 0, 1);
        }

        float accv = 0.f;
#pragma unroll
        for (int n = 0; n < NST; ++n) {
            float dA = expf(dtv * a[n]);
            h[n] = dA * h[n] + dtv * sB[n] * xv;
            accv += h[n] * sC[n];
        }
        float yv = accv + xv * Dd;
        float sg = zv / (1.f + expf(-zv));
        y[row * DINNER + d] = yv * sg;
        __syncthreads();
    }
}

// ---------------------------------------------------------------------------
// ycomb16 = f16(yf + flip(yb))   (feeds the out-projection GEMM only)
// ---------------------------------------------------------------------------
__global__ void combine_y(const float* __restrict__ yf, const float* __restrict__ yb,
                          _Float16* __restrict__ yc16)
{
    int idx = blockIdx.x * blockDim.x + threadIdx.x;
    if (idx >= RTOT * DINNER) return;
    int b   = idx / (LSEQ * DINNER);
    int rem = idx % (LSEQ * DINNER);
    int l   = rem / DINNER;
    int d   = rem % DINNER;
    float v = yf[idx] + yb[((size_t)b * LSEQ + (LSEQ - 1 - l)) * DINNER + d];
    yc16[idx] = (_Float16)v;
}

// ---------------------------------------------------------------------------
// out[b,:] = h[b,POS,:] + MLP(concat(state,action))
// ---------------------------------------------------------------------------
__global__ void head_kernel(const float* __restrict__ hbuf,
                            const float* __restrict__ sv, const float* __restrict__ ac,
                            const float* __restrict__ cw1, const float* __restrict__ cb1,
                            const float* __restrict__ cw2, const float* __restrict__ cb2,
                            float* __restrict__ out)
{
    const int b = blockIdx.x;
    const int d = threadIdx.x;                     // 0..383
    __shared__ float cmd[20];
    __shared__ float t1[DMODEL];
    if (d < 16)      cmd[d] = sv[b * 16 + d];
    else if (d < 20) cmd[d] = ac[b * 4 + d - 16];
    __syncthreads();
    float s = cb1[d];
#pragma unroll
    for (int j = 0; j < 20; ++j) s += cw1[d * 20 + j] * cmd[j];
    t1[d] = fmaxf(s, 0.f);
    __syncthreads();
    float o = cb2[d];
    for (int j = 0; j < DMODEL; ++j) o += cw2[d * DMODEL + j] * t1[j];
    out[(size_t)b * DMODEL + d] =
        hbuf[((size_t)b * LSEQ + POSC) * DMODEL + d] + o;
}

// ---------------------------------------------------------------------------
extern "C" void kernel_launch(void* const* d_in, const int* in_sizes, int n_in,
                              void* d_out, int out_size, void* d_ws, size_t ws_size,
                              hipStream_t stream)
{
    (void)in_sizes; (void)n_in; (void)out_size; (void)ws_size;
    const float* depth_seq = (const float*)d_in[0];
    const float* state_vec = (const float*)d_in[1];
    const float* action    = (const float*)d_in[2];
    const float* patch_w   = (const float*)d_in[3];
    const float* patch_b   = (const float*)d_in[4];
    const float* cls_token = (const float*)d_in[5];
    const float* pos_embed = (const float*)d_in[6];
    const float* ln_w      = (const float*)d_in[7];
    const float* ln_b      = (const float*)d_in[8];
    const float* in_proj_w = (const float*)d_in[9];
    const float* conv_w    = (const float*)d_in[10];
    const float* conv_b    = (const float*)d_in[11];
    const float* conv_wb   = (const float*)d_in[12];
    const float* conv_bb   = (const float*)d_in[13];
    const float* xproj_w   = (const float*)d_in[14];
    const float* xproj_wb  = (const float*)d_in[15];
    const float* dtproj_w  = (const float*)d_in[16];
    const float* dtproj_b  = (const float*)d_in[17];
    const float* dtproj_wb = (const float*)d_in[18];
    const float* dtproj_bb = (const float*)d_in[19];
    const float* A_log     = (const float*)d_in[20];
    const float* A_logb    = (const float*)d_in[21];
    const float* Dp        = (const float*)d_in[22];
    const float* Dpb       = (const float*)d_in[23];
    const float* out_w     = (const float*)d_in[24];
    const float* lnf_w     = (const float*)d_in[25];
    const float* lnf_b     = (const float*)d_in[26];
    const float* cw1       = (const float*)d_in[27];
    const float* cb1       = (const float*)d_in[28];
    const float* cw2       = (const float*)d_in[29];
    const float* cb2       = (const float*)d_in[30];
    float* out = (float*)d_out;

    // ---- f32 arena ----
    float* ws = (float*)d_ws;
    size_t off = 0;
    auto alloc = [&](size_t n) { n = (n + 3) & ~(size_t)3; float* p = ws + off; off += n; return p; };
    float* hidden   = alloc((size_t)RTOT * DMODEL);
    float* residual = alloc((size_t)RTOT * DMODEL);
    float* hbuf     = alloc((size_t)RTOT * DMODEL);
    float* xz       = alloc((size_t)RTOT * 2 * DINNER);
    float* xc_f     = alloc((size_t)RTOT * DINNER);
    float* xc_b     = alloc((size_t)RTOT * DINNER);
    float* xdb_f    = alloc((size_t)RTOT * XDB);
    float* xdb_b    = alloc((size_t)RTOT * XDB);
    float* dt_f     = alloc((size_t)RTOT * DINNER);
    float* dt_b     = alloc((size_t)RTOT * DINNER);
    float* yf       = alloc((size_t)RTOT * DINNER);
    float* yb       = alloc((size_t)RTOT * DINNER);
    float* tokG     = alloc((size_t)BB * NTOKC * DMODEL);

    // ---- f16 arena (16B-aligned: float offsets are multiples of 4) ----
    _Float16* h16base = (_Float16*)(ws + off);
    size_t hoff = 0;
    auto halloc = [&](size_t n) { n = (n + 7) & ~(size_t)7; _Float16* p = h16base + hoff; hoff += n; return p; };
    _Float16* xr16      = halloc((size_t)BB * NTOKC * PATCHK);     // 1024 x 256
    _Float16* patchw16  = halloc((size_t)DMODEL * PATCHK);         // 384 x 256
    _Float16* hbuf16    = halloc((size_t)RPAD * DMODEL);           // 1040 x 384
    _Float16* inprojw16 = halloc((size_t)2 * DINNER * DMODEL);     // 1536 x 384
    _Float16* xcf16     = halloc((size_t)RPAD * DINNER);           // 1040 x 768
    _Float16* xcb16     = halloc((size_t)RPAD * DINNER);
    _Float16* xprojwf16 = halloc((size_t)64 * DINNER);             // 64 x 768
    _Float16* xprojwb16 = halloc((size_t)64 * DINNER);
    _Float16* xdbf16    = halloc((size_t)RPAD * 32);               // 1040 x 32
    _Float16* xdbb16    = halloc((size_t)RPAD * 32);
    _Float16* dtprojwf16= halloc((size_t)DINNER * 32);             // 768 x 32
    _Float16* dtprojwb16= halloc((size_t)DINNER * 32);
    _Float16* ycomb16   = halloc((size_t)RPAD * DINNER);           // 1040 x 768
    _Float16* outw16    = halloc((size_t)DMODEL * DINNER);         // 384 x 768

    // Zero the f16 arena once per call: pad rows of fused producers stay 0.
    {
        int n = (int)hoff;
        zero_h16<<<(n + 255) / 256, 256, 0, stream>>>(h16base, n);
    }

    const dim3 gblk(32, 4);   // 4 waves/block, one 16x64 strip per wave
    auto gemm = [&](const _Float16* A16, const _Float16* W16, const float* bias,
                    float* C, int ldc, int M, int N, int Kp, int Np, int act) {
        int strips = ((M + 15) / 16) * (Np / 64);
        gemm_f16wmma<<<dim3((strips + 3) / 4), gblk, 0, stream>>>(
            A16, W16, bias, C, ldc, M, N, Kp, Np, act);
    };
    auto cvt = [&](const float* src, int rows, int cols, int ldsrc,
                   _Float16* dst, int rowsP, int colsP) {
        int n = rowsP * colsP;
        cvt_pad_f16<<<(n + 255) / 256, 256, 0, stream>>>(src, rows, cols, ldsrc,
                                                         dst, rowsP, colsP);
    };

    // ---- patch embedding ----
    {
        int n = BB * NTOKC * PATCHK;
        gather_xr<<<(n + 255) / 256, 256, 0, stream>>>(depth_seq, xr16);
    }
    cvt(patch_w, DMODEL, PATCHK, PATCHK, patchw16, DMODEL, PATCHK);
    gemm(xr16, patchw16, patch_b, tokG, DMODEL, BB * NTOKC, DMODEL, PATCHK, DMODEL, 0);
    {
        int n = RTOT * DMODEL;
        assemble_tokens<<<(n + 255) / 256, 256, 0, stream>>>(tokG, cls_token,
                                                             pos_embed, hidden);
    }

    // ---- 4 bidirectional Mamba layers ----
    for (int i = 0; i < 4; ++i) {
        resid_ln<<<(RTOT + 3) / 4, dim3(32, 4), 0, stream>>>(
            hidden, residual, hbuf, hbuf16,
            ln_w + i * DMODEL, ln_b + i * DMODEL, i == 0 ? 0 : 1, RTOT);

        cvt(in_proj_w + (size_t)i * 2 * DINNER * DMODEL, 2 * DINNER, DMODEL, DMODEL,
            inprojw16, 2 * DINNER, DMODEL);
        gemm(hbuf16, inprojw16, nullptr, xz, 2 * DINNER,
             RTOT, 2 * DINNER, DMODEL, 2 * DINNER, 0);

        {
            int n = RTOT * DINNER, g = (n + 255) / 256;
            conv_silu<<<g, 256, 0, stream>>>(xz, 2 * DINNER,
                conv_w + (size_t)i * DINNER * KCONV, conv_b + i * DINNER, xc_f, xcf16, 0);
            conv_silu<<<g, 256, 0, stream>>>(xz, 2 * DINNER,
                conv_wb + (size_t)i * DINNER * KCONV, conv_bb + i * DINNER, xc_b, xcb16, 1);
        }

        cvt(xproj_w + (size_t)i * XDB * DINNER, XDB, DINNER, DINNER, xprojwf16, 64, DINNER);
        cvt(xproj_wb + (size_t)i * XDB * DINNER, XDB, DINNER, DINNER, xprojwb16, 64, DINNER);
        gemm(xcf16, xprojwf16, nullptr, xdb_f, XDB, RTOT, XDB, DINNER, 64, 0);
        gemm(xcb16, xprojwb16, nullptr, xdb_b, XDB, RTOT, XDB, DINNER, 64, 0);

        // dt = softplus(xdb[:, :24] @ dtproj^T + bias)
        cvt(xdb_f, RTOT, DTRANK, XDB, xdbf16, RPAD, 32);
        cvt(xdb_b, RTOT, DTRANK, XDB, xdbb16, RPAD, 32);
        cvt(dtproj_w + (size_t)i * DINNER * DTRANK, DINNER, DTRANK, DTRANK,
            dtprojwf16, DINNER, 32);
        cvt(dtproj_wb + (size_t)i * DINNER * DTRANK, DINNER, DTRANK, DTRANK,
            dtprojwb16, DINNER, 32);
        gemm(xdbf16, dtprojwf16, dtproj_b + i * DINNER, dt_f, DINNER,
             RTOT, DINNER, 32, DINNER, 1);
        gemm(xdbb16, dtprojwb16, dtproj_bb + i * DINNER, dt_b, DINNER,
             RTOT, DINNER, 32, DINNER, 1);

        ssm_scan<<<dim3(DINNER / 256, BB), 256, 0, stream>>>(
            xc_f, dt_f, xdb_f, xz + DINNER, 2 * DINNER,
            A_log + (size_t)i * DINNER * NST, Dp + i * DINNER, yf, 0);
        ssm_scan<<<dim3(DINNER / 256, BB), 256, 0, stream>>>(
            xc_b, dt_b, xdb_b, xz + DINNER, 2 * DINNER,
            A_logb + (size_t)i * DINNER * NST, Dpb + i * DINNER, yb, 1);

        {
            int n = RTOT * DINNER;
            combine_y<<<(n + 255) / 256, 256, 0, stream>>>(yf, yb, ycomb16);
        }

        cvt(out_w + (size_t)i * DMODEL * DINNER, DMODEL, DINNER, DINNER,
            outw16, DMODEL, DINNER);
        gemm(ycomb16, outw16, nullptr, hidden, DMODEL, RTOT, DMODEL, DINNER, DMODEL, 0);
    }

    // ---- final LN + head ----
    resid_ln<<<(RTOT + 3) / 4, dim3(32, 4), 0, stream>>>(
        hidden, residual, hbuf, hbuf16, lnf_w, lnf_b, 1, RTOT);
    head_kernel<<<BB, DMODEL, 0, stream>>>(hbuf, state_vec, action,
                                           cw1, cb1, cw2, cb2, out);
}
